// imageCaptionModel_81724637708550
// MI455X (gfx1250) — compile-verified, matrix operand order checked by reading, not run on hardware
//
#include <hip/hip_runtime.h>

typedef __bf16 bf16;
typedef __attribute__((ext_vector_type(16))) __bf16 v16bf;
typedef __attribute__((ext_vector_type(8)))  float  v8f;

#define BB   128
#define SS   20
#define HH   512
#define EE   512
#define VV   10000
#define K0   1536          // d0 = d1 = E+H+H = 2H+H
#define G4   2048          // 4*H gate width
#define KATT 1024          // 2*H
#define KT0  (K0/32)       // 48 k-tiles for gate GEMMs
#define KTA  (KATT/32)     // 32 k-tiles for attention GEMM
#define KTO  (HH/32)       // 16 k-tiles for logits GEMM
#define NTO  (VV/16)       // 625 n-tiles for logits
#define NTO_PAD 640        // padded to multiple of 16 tiles (4 waves x 4 accs)

// ---------- helpers ----------

__device__ inline bf16 f2bf(float f) {
    unsigned u = __float_as_uint(f);
    u += 0x7FFFu + ((u >> 16) & 1u);           // round-to-nearest-even
    unsigned short h = (unsigned short)(u >> 16);
    bf16 r;
    __builtin_memcpy(&r, &h, 2);
    return r;
}

__device__ inline float sigm(float x) { return 1.0f / (1.0f + __expf(-x)); }

// load a 16-value bf16 fragment from two contiguous 16B runs
__device__ inline v16bf ldfrag2(const bf16* lo, const bf16* hi) {
    union { v16bf v; uint4 u[2]; } r;
    r.u[0] = *(const uint4*)lo;
    r.u[1] = *(const uint4*)hi;
    return r.v;
}

// load a packed 32B B-fragment (16 contiguous bf16)
__device__ inline v16bf ldfragp(const bf16* p) {
    union { v16bf v; uint4 u[2]; } r;
    const uint4* q = (const uint4*)p;
    r.u[0] = q[0];
    r.u[1] = q[1];
    return r.v;
}

// ---------- weight packing ----------
// Pack fp32 W (K x N, row-major) into bf16 WMMA B-fragment tile layout:
// tile t = nTile*KT + kTile; within tile: lane l (0..31) holds 16 contiguous
// bf16: e<8 -> K = kTile*32 + (l>>4)*8 + e ; e>=8 -> K = +16 more.
// column n = nTile*16 + (l&15). nTile >= ntReal produces zero padding.
__global__ __launch_bounds__(256) void pack_weight_kernel(
    const float* __restrict__ W, bf16* __restrict__ P, int K, int N,
    int ntReal, int totalPadded) {
    int idx = blockIdx.x * 256 + threadIdx.x;
    if (idx >= totalPadded) return;
    int t  = idx >> 9;          // tile index
    int w  = idx & 511;
    int l  = w >> 4;            // lane
    int e  = w & 15;            // element within fragment
    int KT = K >> 5;
    int nt = t / KT;
    int kt = t - nt * KT;
    float v = 0.0f;
    if (nt < ntReal) {
        int n  = nt * 16 + (l & 15);
        int kh = (l >> 4) << 3;
        int k  = kt * 32 + ((e < 8) ? (kh + e) : (16 + kh + (e - 8)));
        v = W[(size_t)k * N + n];
    }
    P[idx] = f2bf(v);
}

// ---------- state init / finalize ----------

__global__ __launch_bounds__(256) void init_state_kernel(
    const float* __restrict__ init, float* __restrict__ s0, float* __restrict__ s1) {
    int idx = blockIdx.x * 256 + threadIdx.x;
    if (idx >= BB * 1024) return;
    s0[idx] = init[idx];
    s1[idx] = init[BB * 1024 + idx];
}

__global__ __launch_bounds__(256) void final_state_kernel(
    const float* __restrict__ s0, const float* __restrict__ s1, float* __restrict__ out) {
    int idx = blockIdx.x * 256 + threadIdx.x;
    if (idx >= 2 * BB * 1024) return;
    int layer = idx / (BB * 1024);
    int rem   = idx - layer * (BB * 1024);
    out[(size_t)BB * SS * VV + idx] = layer ? s1[rem] : s0[rem];
}

// ---------- activation packing (fp32 concats -> bf16 row-major) ----------

__global__ __launch_bounds__(256) void pack_a0_kernel(
    const int* __restrict__ tokens, const float* __restrict__ img,
    const float* __restrict__ emb, const float* __restrict__ s0,
    bf16* __restrict__ A0, int s) {
    int idx = blockIdx.x * 256 + threadIdx.x;
    if (idx >= BB * K0) return;
    int m = idx / K0, k = idx - m * K0;
    float v;
    if (k < HH)            v = img[m * HH + k];
    else if (k < HH + EE) { int tok = tokens[m * SS + s];
                            v = emb[(size_t)tok * EE + (k - HH)]; }
    else                   v = s0[m * 1024 + (k - (HH + EE))];   // h0 prev
    A0[idx] = f2bf(v);
}

__global__ __launch_bounds__(256) void pack_aatt_kernel(
    const float* __restrict__ s0, bf16* __restrict__ Aatt) {
    int idx = blockIdx.x * 256 + threadIdx.x;
    if (idx >= BB * KATT) return;
    Aatt[idx] = f2bf(s0[idx]);                                   // full [h0|c0]
}

__global__ __launch_bounds__(256) void pack_a1_kernel(
    const float* __restrict__ s0, const float* __restrict__ att,
    const float* __restrict__ s1, bf16* __restrict__ A1) {
    int idx = blockIdx.x * 256 + threadIdx.x;
    if (idx >= BB * K0) return;
    int m = idx / K0, k = idx - m * K0;
    float v;
    if (k < HH)        v = s0[m * 1024 + k];                     // h0 new
    else if (k < 1024) v = att[m * HH + (k - HH)];               // attention
    else               v = s1[m * 1024 + (k - 1024)];            // h1 prev
    A1[idx] = f2bf(v);
}

__global__ __launch_bounds__(256) void pack_aout_kernel(
    const float* __restrict__ s1, bf16* __restrict__ Aout) {
    int idx = blockIdx.x * 256 + threadIdx.x;
    if (idx >= BB * HH) return;
    int m = idx >> 9, j = idx & 511;
    Aout[idx] = f2bf(s1[m * 1024 + j]);                          // h1 new
}

// ---------- fused gates GEMM + LSTM cell ----------
// grid (BB/16, HH/32), block 128 (4 waves). Wave w computes TWO 16x16 tiles of
// gate group w (i/f/g/o): j-tiles jt0=blockIdx.y*2, jt0+1. One A fragment is
// reused against both B fragments per k-step. state (BB x 1024) = [h | c],
// updated in place.
__global__ __launch_bounds__(128) void gemm_lstm_kernel(
    const bf16* __restrict__ A, const bf16* __restrict__ P,
    const float* __restrict__ bias, float* __restrict__ state) {
    const int lane  = threadIdx.x & 31;
    const int wave  = threadIdx.x >> 5;
    const int mBase = blockIdx.x * 16;
    const int jt0   = blockIdx.y * 2;             // 0..30 (step 2)
    const int nT0   = wave * (HH / 16) + jt0;     // tile within 2048 cols
    const int nT1   = nT0 + 1;

    v8f acc0 = {}, acc1 = {};
    const bf16* prow0 = P + ((size_t)nT0 * KT0) * 512 + lane * 16;
    const bf16* prow1 = P + ((size_t)nT1 * KT0) * 512 + lane * 16;
    const int   m     = mBase + (lane & 15);
    const bf16* arow  = A + (size_t)m * K0 + ((lane >> 4) << 3);
    #pragma unroll 4
    for (int kt = 0; kt < KT0; ++kt) {
        v16bf a  = ldfrag2(arow + kt * 32, arow + kt * 32 + 16);
        v16bf b0 = ldfragp(prow0 + (size_t)kt * 512);
        v16bf b1 = ldfragp(prow1 + (size_t)kt * 512);
        acc0 = __builtin_amdgcn_wmma_f32_16x16x32_bf16(
                   false, a, false, b0, (short)0, acc0, false, false);
        acc1 = __builtin_amdgcn_wmma_f32_16x16x32_bf16(
                   false, a, false, b1, (short)0, acc1, false, false);
    }

    __shared__ float gl[4][16][32];
    const float bv0  = bias[nT0 * 16 + (lane & 15)];
    const float bv1  = bias[nT1 * 16 + (lane & 15)];
    const int   mrow = (lane >> 4) * 8;
    #pragma unroll
    for (int i = 0; i < 8; ++i) {
        gl[wave][mrow + i][(lane & 15)]      = acc0[i] + bv0;
        gl[wave][mrow + i][16 + (lane & 15)] = acc1[i] + bv1;
    }
    __syncthreads();

    #pragma unroll
    for (int i = 0; i < 4; ++i) {
        int e  = threadIdx.x + i * 128;           // 512 outputs
        int ml = e >> 5, jl = e & 31;
        float gi = gl[0][ml][jl];
        float gf = gl[1][ml][jl];
        float gg = gl[2][ml][jl];
        float go = gl[3][ml][jl];
        int mm = mBase + ml;
        int j  = blockIdx.y * 32 + jl;
        float cprev = state[mm * 1024 + HH + j];
        float cn = sigm(gf) * cprev + sigm(gi) * tanhf(gg);
        float hn = sigm(go) * tanhf(cn);
        state[mm * 1024 + j]      = hn;
        state[mm * 1024 + HH + j] = cn;
    }
}

// ---------- attention GEMM: att = tanh(s0 @ W_att + b_att) ----------
// grid (BB/16, 4), block 128; wave handles 2 consecutive n-tiles (NT=32)
__global__ __launch_bounds__(128) void gemm_att_kernel(
    const bf16* __restrict__ A, const bf16* __restrict__ P,
    const float* __restrict__ bias, float* __restrict__ att) {
    const int lane  = threadIdx.x & 31;
    const int wave  = threadIdx.x >> 5;
    const int mBase = blockIdx.x * 16;
    const int nT0   = (blockIdx.y * 4 + wave) * 2;
    const int nT1   = nT0 + 1;

    v8f acc0 = {}, acc1 = {};
    const bf16* prow0 = P + ((size_t)nT0 * KTA) * 512 + lane * 16;
    const bf16* prow1 = P + ((size_t)nT1 * KTA) * 512 + lane * 16;
    const int   m     = mBase + (lane & 15);
    const bf16* arow  = A + (size_t)m * KATT + ((lane >> 4) << 3);
    #pragma unroll 4
    for (int kt = 0; kt < KTA; ++kt) {
        v16bf a  = ldfrag2(arow + kt * 32, arow + kt * 32 + 16);
        v16bf b0 = ldfragp(prow0 + (size_t)kt * 512);
        v16bf b1 = ldfragp(prow1 + (size_t)kt * 512);
        acc0 = __builtin_amdgcn_wmma_f32_16x16x32_bf16(
                   false, a, false, b0, (short)0, acc0, false, false);
        acc1 = __builtin_amdgcn_wmma_f32_16x16x32_bf16(
                   false, a, false, b1, (short)0, acc1, false, false);
    }

    const float bv0  = bias[nT0 * 16 + (lane & 15)];
    const float bv1  = bias[nT1 * 16 + (lane & 15)];
    const int   mrow = (lane >> 4) * 8;
    const int   n0   = nT0 * 16 + (lane & 15);
    const int   n1   = nT1 * 16 + (lane & 15);
    #pragma unroll
    for (int i = 0; i < 8; ++i) {
        att[(mBase + mrow + i) * HH + n0] = tanhf(acc0[i] + bv0);
        att[(mBase + mrow + i) * HH + n1] = tanhf(acc1[i] + bv1);
    }
}

// ---------- logits GEMM: out[b, s, :] = h1 @ W_out + b_out ----------
// grid (BB/16, NTO_PAD/16), block 128; wave handles 4 consecutive n-tiles.
// P is padded to NTO_PAD tiles so the K loop never branches; only stores are
// guarded (wave-uniform), keeping EXEC all-ones at every WMMA.
__global__ __launch_bounds__(128) void gemm_logits_kernel(
    const bf16* __restrict__ A, const bf16* __restrict__ P,
    const float* __restrict__ bias, float* __restrict__ out, int s) {
    const int lane  = threadIdx.x & 31;
    const int wave  = threadIdx.x >> 5;
    const int mBase = blockIdx.x * 16;
    const int nT0   = (blockIdx.y * 4 + wave) * 4;

    v8f acc0 = {}, acc1 = {}, acc2 = {}, acc3 = {};
    const bf16* prow0 = P + ((size_t)(nT0 + 0) * KTO) * 512 + lane * 16;
    const bf16* prow1 = P + ((size_t)(nT0 + 1) * KTO) * 512 + lane * 16;
    const bf16* prow2 = P + ((size_t)(nT0 + 2) * KTO) * 512 + lane * 16;
    const bf16* prow3 = P + ((size_t)(nT0 + 3) * KTO) * 512 + lane * 16;
    const int   m     = mBase + (lane & 15);
    const bf16* arow  = A + (size_t)m * HH + ((lane >> 4) << 3);
    #pragma unroll 4
    for (int kt = 0; kt < KTO; ++kt) {
        v16bf a  = ldfrag2(arow + kt * 32, arow + kt * 32 + 16);
        v16bf b0 = ldfragp(prow0 + (size_t)kt * 512);
        v16bf b1 = ldfragp(prow1 + (size_t)kt * 512);
        v16bf b2 = ldfragp(prow2 + (size_t)kt * 512);
        v16bf b3 = ldfragp(prow3 + (size_t)kt * 512);
        acc0 = __builtin_amdgcn_wmma_f32_16x16x32_bf16(
                   false, a, false, b0, (short)0, acc0, false, false);
        acc1 = __builtin_amdgcn_wmma_f32_16x16x32_bf16(
                   false, a, false, b1, (short)0, acc1, false, false);
        acc2 = __builtin_amdgcn_wmma_f32_16x16x32_bf16(
                   false, a, false, b2, (short)0, acc2, false, false);
        acc3 = __builtin_amdgcn_wmma_f32_16x16x32_bf16(
                   false, a, false, b3, (short)0, acc3, false, false);
    }

    const int mrow = (lane >> 4) * 8;
    const int ncol = lane & 15;
    v8f accs[4] = {acc0, acc1, acc2, acc3};
    #pragma unroll
    for (int t = 0; t < 4; ++t) {
        int nT = nT0 + t;
        if (nT < NTO) {                           // wave-uniform store guard
            float bv = bias[nT * 16 + ncol];
            int   n  = nT * 16 + ncol;
            #pragma unroll
            for (int i = 0; i < 8; ++i)
                out[(size_t)(mBase + mrow + i) * (SS * VV) + (size_t)s * VV + n]
                    = accs[t][i] + bv;
        }
    }
}

// ---------- host ----------

extern "C" void kernel_launch(void* const* d_in, const int* in_sizes, int n_in,
                              void* d_out, int out_size, void* d_ws, size_t ws_size,
                              hipStream_t stream) {
    const int*   xTokens = (const int*)  d_in[0];
    const float* baseimg = (const float*)d_in[1];
    const float* initst  = (const float*)d_in[2];
    const float* emb     = (const float*)d_in[3];
    const float* W0      = (const float*)d_in[4];
    const float* b0      = (const float*)d_in[5];
    const float* W1      = (const float*)d_in[6];
    const float* b1      = (const float*)d_in[7];
    const float* Watt    = (const float*)d_in[8];
    const float* batt    = (const float*)d_in[9];
    const float* Wout    = (const float*)d_in[10];
    const float* bout    = (const float*)d_in[11];
    float* out = (float*)d_out;

    char* ws = (char*)d_ws;
    size_t off = 0;
    auto alloc = [&](size_t bytes) -> void* {
        void* p = ws + off;
        off = (off + bytes + 255) & ~(size_t)255;
        return p;
    };
    const int woutPadded = NTO_PAD * KTO * 512;           // padded elements
    bf16* pW0   = (bf16*)alloc((size_t)K0 * G4 * 2);
    bf16* pW1   = (bf16*)alloc((size_t)K0 * G4 * 2);
    bf16* pWatt = (bf16*)alloc((size_t)KATT * HH * 2);
    bf16* pWout = (bf16*)alloc((size_t)woutPadded * 2);
    bf16* A0    = (bf16*)alloc((size_t)BB * K0 * 2);
    bf16* Aatt  = (bf16*)alloc((size_t)BB * KATT * 2);
    bf16* A1    = (bf16*)alloc((size_t)BB * K0 * 2);
    bf16* Aout  = (bf16*)alloc((size_t)BB * HH * 2);
    float* s0   = (float*)alloc((size_t)BB * 1024 * 4);
    float* s1   = (float*)alloc((size_t)BB * 1024 * 4);
    float* att  = (float*)alloc((size_t)BB * HH * 4);

    // one-time packs (re-done every call for determinism)
    pack_weight_kernel<<<(K0 * G4 + 255) / 256, 256, 0, stream>>>(
        W0, pW0, K0, G4, G4 / 16, K0 * G4);
    pack_weight_kernel<<<(K0 * G4 + 255) / 256, 256, 0, stream>>>(
        W1, pW1, K0, G4, G4 / 16, K0 * G4);
    pack_weight_kernel<<<(KATT * HH + 255) / 256, 256, 0, stream>>>(
        Watt, pWatt, KATT, HH, HH / 16, KATT * HH);
    pack_weight_kernel<<<(woutPadded + 255) / 256, 256, 0, stream>>>(
        Wout, pWout, HH, VV, NTO, woutPadded);
    init_state_kernel<<<(BB * 1024 + 255) / 256, 256, 0, stream>>>(initst, s0, s1);

    dim3 gGate(BB / 16, HH / 32);           // 8 x 16, 2 j-tiles per wave
    dim3 gAtt(BB / 16, 4);                  // 8 x 4,  2 n-tiles per wave
    dim3 gOut(BB / 16, NTO_PAD / 16);       // 8 x 40, 4 n-tiles per wave

    for (int s = 0; s < SS; ++s) {
        pack_a0_kernel<<<(BB * K0 + 255) / 256, 256, 0, stream>>>(xTokens, baseimg, emb, s0, A0, s);
        gemm_lstm_kernel<<<gGate, 128, 0, stream>>>(A0, pW0, b0, s0);
        pack_aatt_kernel<<<(BB * KATT + 255) / 256, 256, 0, stream>>>(s0, Aatt);
        gemm_att_kernel<<<gAtt, 128, 0, stream>>>(Aatt, pWatt, batt, att);
        pack_a1_kernel<<<(BB * K0 + 255) / 256, 256, 0, stream>>>(s0, att, s1, A1);
        gemm_lstm_kernel<<<gGate, 128, 0, stream>>>(A1, pW1, b1, s1);
        pack_aout_kernel<<<(BB * HH + 255) / 256, 256, 0, stream>>>(s1, Aout);
        gemm_logits_kernel<<<gOut, 128, 0, stream>>>(Aout, pWout, bout, out, s);
    }
    final_state_kernel<<<(2 * BB * 1024 + 255) / 256, 256, 0, stream>>>(s0, s1, out);
}
